// OGRENet_21827023798512
// MI455X (gfx1250) — compile-verified, matrix operand order checked by reading, not run on hardware
//
#include <hip/hip_runtime.h>
#include <hip/hip_fp16.h>

typedef __attribute__((ext_vector_type(16))) _Float16 v16h;
typedef __attribute__((ext_vector_type(8)))  _Float16 v8h;
typedef __attribute__((ext_vector_type(8)))  float    v8f;

#define NN   20000
#define NE   160000
#define NG   64
#define XD   9
#define URAW 4096
#define UD   256
#define EINR 275   // real edge-MLP input width
#define EINP 320   // padded to multiple of 64
#define EH   1024
#define EO   512
#define N1R  521
#define N1P  576
#define NH   512
#define N2R  777
#define N2P  832

#define WMMA16(a, b, c) __builtin_amdgcn_wmma_f32_16x16x32_f16( \
    false, (a), false, (b), (short)0, (c), false, false)

#define LDT   72            // LDS row stride (halves): 144B, 16B-aligned, bank-spread
#define TILEH (128 * LDT)

// ---------------------------------------------------------------------------
// Tiled GEMM: C[M,N] = act( A[M,Kpad](f16) * Wt[N,Kpad](f16, pre-transposed) + bias )
// BM=BN=128, BK=64, 256 threads (8 wave32), wave tile 32x64, 16 WMMA / K-step.
// Register-staged software pipeline: next tile's global loads issue before compute.
// ---------------------------------------------------------------------------
__global__ __launch_bounds__(256) void gemm_f16_wmma(
    const _Float16* __restrict__ A, const _Float16* __restrict__ Wt,
    const float* __restrict__ bias, _Float16* __restrict__ C,
    int M, int Kpad, int N, int relu)
{
  __shared__ __align__(16) _Float16 As[TILEH];
  __shared__ __align__(16) _Float16 Bs[TILEH];

  const int bm   = blockIdx.x * 128;
  const int bn   = blockIdx.y * 128;
  const int lane = threadIdx.x & 31;
  const int wid  = threadIdx.x >> 5;
  const int wm   = (wid >> 1) * 32;    // 4 waves along M
  const int wn   = (wid & 1) * 64;     // 2 waves along N
  const int lh   = lane & 15;
  const int ls   = lane >> 4;

  const int sr = threadIdx.x >> 3;        // staging row 0..31 (+32 per rep)
  const int sc = (threadIdx.x & 7) * 8;   // staging chunk (halves)

  v8f acc[2][4] = {};
  v8h ar[4], br[4];

  // load one 128x64 A tile + 128x64 Wt tile into registers
  auto ldregs = [&](int k0) {
    #pragma unroll
    for (int i = 0; i < 4; ++i) {
      int r = sr + i * 32;
      v8h v = {};
      if (bm + r < M) v = *(const v8h*)(A + (size_t)(bm + r) * Kpad + k0 + sc);
      ar[i] = v;
      br[i] = *(const v8h*)(Wt + (size_t)(bn + r) * Kpad + k0 + sc);
    }
  };
  auto stlds = [&]() {
    #pragma unroll
    for (int i = 0; i < 4; ++i) {
      int r = sr + i * 32;
      *(v8h*)&As[r * LDT + sc] = ar[i];
      *(v8h*)&Bs[r * LDT + sc] = br[i];
    }
  };

  const int nk = Kpad >> 6;
  ldregs(0);
  for (int i = 0; i < nk; ++i) {
    stlds();
    __syncthreads();
    if (i + 1 < nk) ldregs((i + 1) << 6);   // overlaps with WMMA below

    #pragma unroll
    for (int s = 0; s < 2; ++s) {           // two K=32 slices of the 64-wide tile
      const int kb = s * 32 + ls * 16;
      v16h a[2];
      #pragma unroll
      for (int sm = 0; sm < 2; ++sm) {
        const _Float16* p = &As[(wm + sm * 16 + lh) * LDT + kb];
        ((v8h*)&a[sm])[0] = *(const v8h*)p;
        ((v8h*)&a[sm])[1] = *(const v8h*)(p + 8);
      }
      #pragma unroll
      for (int sn = 0; sn < 4; ++sn) {
        const _Float16* p = &Bs[(wn + sn * 16 + lh) * LDT + kb];
        v16h b;
        ((v8h*)&b)[0] = *(const v8h*)p;
        ((v8h*)&b)[1] = *(const v8h*)(p + 8);
        acc[0][sn] = WMMA16(a[0], b, acc[0][sn]);
        acc[1][sn] = WMMA16(a[1], b, acc[1][sn]);
      }
    }
    __syncthreads();
  }

  // epilogue: bias + optional ReLU, store f16
  #pragma unroll
  for (int sm = 0; sm < 2; ++sm)
    #pragma unroll
    for (int sn = 0; sn < 4; ++sn) {
      int gn = bn + wn + sn * 16 + lh;
      float bb = bias[gn];
      #pragma unroll
      for (int r = 0; r < 8; ++r) {
        int gm = bm + wm + sm * 16 + ls * 8 + r;
        if (gm < M) {
          float v = acc[sm][sn][r] + bb;
          if (relu) v = fmaxf(v, 0.f);
          C[(size_t)gm * N + gn] = (_Float16)v;
        }
      }
    }
}

// one-time: W[Kreal,N] f32 -> Wt[N,Kpad] f16 (transposed, zero-padded K)
__global__ void convert_w(const float* __restrict__ W, _Float16* __restrict__ Wt,
                          int Kreal, int Kpad, int N)
{
  int idx = blockIdx.x * 256 + threadIdx.x;
  if (idx >= N * Kpad) return;
  int n = idx / Kpad, k = idx % Kpad;
  float v = (k < Kreal) ? W[(size_t)k * N + n] : 0.f;
  Wt[idx] = (_Float16)v;
}

// u2 = u @ Wsel + bsel -> f16   (64x4096 @ 4096x256, tiny)
__global__ void u2_kernel(const float* __restrict__ u, const float* __restrict__ Wsel,
                          const float* __restrict__ bsel, _Float16* __restrict__ u2h)
{
  int t = blockIdx.x * 256 + threadIdx.x;
  int g = t >> 8, c = t & 255;
  const float* ur = u + (size_t)g * URAW;
  float s = bsel[c];
  for (int k = 0; k < URAW; ++k) s += ur[k] * Wsel[(size_t)k * UD + c];
  u2h[t] = (_Float16)s;
}

// Ein[e] = [x[row] (9) | x[col] (9) | edge_attr (1) | u2[batch[row]] (256) | pad]
__global__ __launch_bounds__(EINP) void build_edge_in(
    const float* __restrict__ x, const float* __restrict__ ea,
    const int* __restrict__ row, const int* __restrict__ col,
    const int* __restrict__ batch, const _Float16* __restrict__ u2h,
    _Float16* __restrict__ Ein)
{
  int e = blockIdx.x, j = threadIdx.x;
  int r = row[e];
  _Float16 v = (_Float16)0.f;
  if (j < 9)        v = (_Float16)x[r * XD + j];
  else if (j < 18)  v = (_Float16)x[col[e] * XD + (j - 9)];
  else if (j == 18) v = (_Float16)ea[e];
  else if (j < EINR) v = u2h[batch[r] * UD + (j - 19)];
  Ein[(size_t)e * EINP + j] = v;
}

// Hin[e] = [x[col] (9) | e_out (512) | pad]
__global__ __launch_bounds__(N1P) void build_node1_in(
    const float* __restrict__ x, const int* __restrict__ col,
    const _Float16* __restrict__ eout, _Float16* __restrict__ Hin)
{
  int e = blockIdx.x, j = threadIdx.x;
  _Float16 v = (_Float16)0.f;
  if (j < 9)        v = (_Float16)x[col[e] * XD + j];
  else if (j < N1R) v = eout[(size_t)e * EO + (j - 9)];
  Hin[(size_t)e * N1P + j] = v;
}

__global__ void scatter_h(const _Float16* __restrict__ h, const int* __restrict__ row,
                          float* __restrict__ agg)
{
  int idx = blockIdx.x * 256 + threadIdx.x;
  if (idx >= NE * NH) return;
  int e = idx >> 9, c = idx & 511;
  atomicAdd(&agg[(size_t)row[e] * NH + c], (float)h[idx]);
}

__global__ void count_rows(const int* __restrict__ row, float* __restrict__ cnt)
{
  int e = blockIdx.x * 256 + threadIdx.x;
  if (e < NE) atomicAdd(&cnt[row[e]], 1.f);
}

// Zin[n] = [x (9) | agg/cnt (512) | u2[batch] (256) | pad]
__global__ __launch_bounds__(N2P) void build_node2_in(
    const float* __restrict__ x, const float* __restrict__ agg,
    const float* __restrict__ cnt, const _Float16* __restrict__ u2h,
    const int* __restrict__ batch, _Float16* __restrict__ Zin)
{
  int n = blockIdx.x, j = threadIdx.x;
  _Float16 v = (_Float16)0.f;
  if (j < 9)        v = (_Float16)x[n * XD + j];
  else if (j < 521) v = (_Float16)(agg[(size_t)n * NH + (j - 9)] / fmaxf(cnt[n], 1.f));
  else if (j < N2R) v = u2h[batch[n] * UD + (j - 521)];
  Zin[(size_t)n * N2P + j] = v;
}

__global__ void final_out(const _Float16* __restrict__ z1, const float* __restrict__ W2,
                          const float* __restrict__ b2, float* __restrict__ out)
{
  int n = blockIdx.x * 256 + threadIdx.x;
  if (n >= NN) return;
  const v8h* zp = (const v8h*)(z1 + (size_t)n * NH);
  float s = 0.f;
  for (int c = 0; c < NH / 8; ++c) {
    v8h z = zp[c];
    #pragma unroll
    for (int j = 0; j < 8; ++j) s += (float)z[j] * W2[c * 8 + j];
  }
  out[n] = s + b2[0];
}

// ---------------------------------------------------------------------------
extern "C" void kernel_launch(void* const* d_in, const int* in_sizes, int n_in,
                              void* d_out, int out_size, void* d_ws, size_t ws_size,
                              hipStream_t stream)
{
  (void)in_sizes; (void)n_in; (void)out_size; (void)ws_size;
  const float* x    = (const float*)d_in[0];
  const float* ea   = (const float*)d_in[1];
  const float* u    = (const float*)d_in[2];
  const int*   ei   = (const int*)d_in[3];
  const int*   bt   = (const int*)d_in[4];
  const float* Wsel = (const float*)d_in[5];
  const float* bsel = (const float*)d_in[6];
  const float* eW1  = (const float*)d_in[7];  const float* eb1 = (const float*)d_in[8];
  const float* eW2  = (const float*)d_in[9];  const float* eb2 = (const float*)d_in[10];
  const float* eW3  = (const float*)d_in[11]; const float* eb3 = (const float*)d_in[12];
  const float* eW4  = (const float*)d_in[13]; const float* eb4 = (const float*)d_in[14];
  const float* eW5  = (const float*)d_in[15]; const float* eb5 = (const float*)d_in[16];
  const float* n1W1 = (const float*)d_in[17]; const float* n1b1 = (const float*)d_in[18];
  const float* n1W2 = (const float*)d_in[19]; const float* n1b2 = (const float*)d_in[20];
  const float* n2W1 = (const float*)d_in[21]; const float* n2b1 = (const float*)d_in[22];
  const float* n2W2 = (const float*)d_in[23]; const float* n2b2 = (const float*)d_in[24];

  const int* row = ei;
  const int* col = ei + NE;

  char* ws = (char*)d_ws;
  size_t off = 0;
  auto alloc = [&](size_t b) { size_t p = off; off += (b + 255) & ~(size_t)255; return p; };
  _Float16* buf0 = (_Float16*)(ws + alloc((size_t)NE * EH * 2));   // 327.7 MB
  _Float16* buf1 = (_Float16*)(ws + alloc((size_t)NE * EH * 2));   // 327.7 MB
  _Float16* u2h  = (_Float16*)(ws + alloc((size_t)NG * UD * 2));
  float*    agg  = (float*)   (ws + alloc((size_t)NN * NH * 4));   // 41 MB
  float*    cnt  = (float*)   (ws + alloc((size_t)NN * 4));
  _Float16* eW1t  = (_Float16*)(ws + alloc((size_t)EH * EINP * 2));
  _Float16* eW2t  = (_Float16*)(ws + alloc((size_t)EH * EH * 2));
  _Float16* eW3t  = (_Float16*)(ws + alloc((size_t)EH * EH * 2));
  _Float16* eW4t  = (_Float16*)(ws + alloc((size_t)EH * EH * 2));
  _Float16* eW5t  = (_Float16*)(ws + alloc((size_t)EO * EH * 2));
  _Float16* n1W1t = (_Float16*)(ws + alloc((size_t)NH * N1P * 2));
  _Float16* n1W2t = (_Float16*)(ws + alloc((size_t)NH * NH * 2));
  _Float16* n2W1t = (_Float16*)(ws + alloc((size_t)NH * N2P * 2));

  dim3 blk(256);
  auto conv = [&](const float* W, _Float16* Wt, int Kreal, int Kpad, int N) {
    convert_w<<<(N * Kpad + 255) / 256, blk, 0, stream>>>(W, Wt, Kreal, Kpad, N);
  };
  auto gemm = [&](const _Float16* A, const _Float16* Wt, const float* b, _Float16* C,
                  int M, int Kpad, int N, int relu) {
    dim3 grid((unsigned)((M + 127) / 128), (unsigned)(N / 128));
    gemm_f16_wmma<<<grid, blk, 0, stream>>>(A, Wt, b, C, M, Kpad, N, relu);
  };

  // one-time weight conversions (f32 -> f16, transposed, K zero-padded)
  conv(eW1, eW1t, EINR, EINP, EH);
  conv(eW2, eW2t, EH, EH, EH);
  conv(eW3, eW3t, EH, EH, EH);
  conv(eW4, eW4t, EH, EH, EH);
  conv(eW5, eW5t, EH, EH, EO);
  conv(n1W1, n1W1t, N1R, N1P, NH);
  conv(n1W2, n1W2t, NH, NH, NH);
  conv(n2W1, n2W1t, N2R, N2P, NH);

  u2_kernel<<<64, blk, 0, stream>>>(u, Wsel, bsel, u2h);

  // Edge MLP (inputs built in buf1, ping-pong through buf0/buf1)
  build_edge_in<<<NE, EINP, 0, stream>>>(x, ea, row, col, bt, u2h, buf1);
  gemm(buf1, eW1t, eb1, buf0, NE, EINP, EH, 1);
  gemm(buf0, eW2t, eb2, buf1, NE, EH, EH, 1);
  gemm(buf1, eW3t, eb3, buf0, NE, EH, EH, 1);
  gemm(buf0, eW4t, eb4, buf1, NE, EH, EH, 1);
  gemm(buf1, eW5t, eb5, buf0, NE, EH, EO, 0);        // e_out = buf0 [NE,512]

  // Node MLP1 over edges
  build_node1_in<<<NE, N1P, 0, stream>>>(x, col, buf0, buf1);
  gemm(buf1, n1W1t, n1b1, buf0, NE, N1P, NH, 1);
  gemm(buf0, n1W2t, n1b2, buf1, NE, NH, NH, 1);      // h = buf1 [NE,512]

  // scatter-mean over destination rows
  hipMemsetAsync(agg, 0, (size_t)NN * NH * 4, stream);
  hipMemsetAsync(cnt, 0, (size_t)NN * 4, stream);
  scatter_h<<<(NE * NH + 255) / 256, blk, 0, stream>>>(buf1, row, agg);
  count_rows<<<(NE + 255) / 256, blk, 0, stream>>>(row, cnt);

  // Node MLP2
  build_node2_in<<<NN, N2P, 0, stream>>>(x, agg, cnt, u2h, bt, buf0);
  gemm(buf0, n2W1t, n2b1, buf1, NN, N2P, NH, 1);     // z1 = buf1 [NN,512]
  final_out<<<(NN + 255) / 256, blk, 0, stream>>>(buf1, n2W2, n2b2, (float*)d_out);
}
